// SimpleAttentionIdentity_22204980920987
// MI455X (gfx1250) — compile-verified, matrix operand order checked by reading, not run on hardware
//
#include <hip/hip_runtime.h>

// ---------------------------------------------------------------------------
// SimpleAttentionIdentity for MI455X (gfx1250): bf16 WMMA pipeline.
//   q,k,v = x @ W{q,k,v}^T ; attn = softmax(q k^T) ; y = attn v + x ; y /= ||y||
// GEMMs: 512-thread blocks (16 waves), 4 subtiles/wave -> 16x1024 per block.
// A-panel (16xK) async-staged to LDS in the prologue (barrier-free K loop);
// B streamed direct from global (contiguous rows by construction).
// ---------------------------------------------------------------------------

typedef __bf16 bf16;
typedef __attribute__((ext_vector_type(16))) __bf16 v16bf;
typedef __attribute__((ext_vector_type(8)))  __bf16 v8bf;
typedef __attribute__((ext_vector_type(8)))  float  v8f;
typedef __attribute__((ext_vector_type(4)))  int    v4i;

static constexpr int kB    = 4;
static constexpr int kS    = 4096;
static constexpr int kD    = 1024;
static constexpr int kRows = kB * kS;          // 16384

// ---- workspace byte offsets -------------------------------------------------
static constexpr size_t XB_OFF = 0;                                   // x bf16   [16384,1024]
static constexpr size_t WB_OFF = XB_OFF + (size_t)kRows * kD * 2;     // W bf16   [3,1024,1024]
static constexpr size_t QB_OFF = WB_OFF + (size_t)3 * kD * kD * 2;    // q bf16   [16384,1024]
static constexpr size_t KB_OFF = QB_OFF + (size_t)kRows * kD * 2;     // k bf16   [16384,1024]
static constexpr size_t VT_OFF = KB_OFF + (size_t)kRows * kD * 2;     // v^T bf16 [4,1024,4096]
static constexpr size_t SC_OFF = VT_OFF + (size_t)kB * kD * kS * 2;   // scores/P [4,4096,4096] bf16
static constexpr size_t ST_OFF = SC_OFF + (size_t)kB * kS * kS * 2;   // stats f32 [16384,2]

// ---- gfx1250 async global->LDS availability ---------------------------------
#if defined(__HIP_DEVICE_COMPILE__) && defined(__AMDGCN__) && \
    __has_builtin(__builtin_amdgcn_global_load_async_to_lds_b128) && \
    __has_builtin(__builtin_amdgcn_s_wait_asynccnt)
#define HAVE_ASYNC_LDS 1
#else
#define HAVE_ASYNC_LDS 0
#endif

#define AS1 __attribute__((address_space(1)))
#define AS3 __attribute__((address_space(3)))

// ---- bf16 16x16x32 fragment helpers -----------------------------------------
// 16-bit A (16xK=32) per-lane layout: lanes 0-15 -> K 0..7 & 16..23,
// lanes 16-31 -> K 8..15 & 24..31 (ISA 7.12.2).  B fragment for column-major
// access mirrors this when the source "column" is a contiguous row in memory.
__device__ __forceinline__ v16bf frag16x32(const bf16* __restrict__ row, int k0, int lane) {
  const int kb = (lane >> 4) << 3;  // 0 or 8
  v8bf lo = *reinterpret_cast<const v8bf*>(row + k0 + kb);
  v8bf hi = *reinterpret_cast<const v8bf*>(row + k0 + kb + 16);
  return __builtin_shufflevector(lo, hi, 0,1,2,3,4,5,6,7,8,9,10,11,12,13,14,15);
}

// A fragment from an LDS-resident 16x1024 panel (row-major, 1024 elem/row).
__device__ __forceinline__ v16bf frag_lds(const bf16* lbuf, int k0, int lane) {
  const int kb = (lane >> 4) << 3;
  const bf16* r = lbuf + (lane & 15) * 1024 + k0 + kb;
  v8bf lo = *reinterpret_cast<const v8bf*>(r);
  v8bf hi = *reinterpret_cast<const v8bf*>(r + 16);
  return __builtin_shufflevector(lo, hi, 0,1,2,3,4,5,6,7,8,9,10,11,12,13,14,15);
}

__device__ __forceinline__ v8f wmma_bf16(v16bf a, v16bf b, v8f c) {
  return __builtin_amdgcn_wmma_f32_16x16x32_bf16(false, a, false, b, (short)0, c, false, false);
}

// Stage a full 16x1024 bf16 panel (row stride `stride` elems) into a 32KB LDS
// buffer. All 512 threads participate: 64B per thread (4x b128 async copies).
__device__ __forceinline__ void stage_panel(const bf16* __restrict__ gbase, int stride,
                                            bf16* lbuf, int tid) {
  const int row = tid >> 5;            // 0..15
  const int off = (tid & 31) * 32;     // element offset within row
  const bf16* g = gbase + (size_t)row * stride + off;
  bf16* l = lbuf + row * 1024 + off;
#if HAVE_ASYNC_LDS
  v4i* gq = (v4i*)g;                   // reinterpret in generic AS (drops const)
  v4i* lq = (v4i*)l;
  __builtin_amdgcn_global_load_async_to_lds_b128((AS1 v4i*)gq, (AS3 v4i*)lq, 0, 0);
  __builtin_amdgcn_global_load_async_to_lds_b128((AS1 v4i*)gq, (AS3 v4i*)lq, 16, 0);
  __builtin_amdgcn_global_load_async_to_lds_b128((AS1 v4i*)gq, (AS3 v4i*)lq, 32, 0);
  __builtin_amdgcn_global_load_async_to_lds_b128((AS1 v4i*)gq, (AS3 v4i*)lq, 48, 0);
#else
  v8bf t0 = *reinterpret_cast<const v8bf*>(g);
  v8bf t1 = *reinterpret_cast<const v8bf*>(g + 8);
  v8bf t2 = *reinterpret_cast<const v8bf*>(g + 16);
  v8bf t3 = *reinterpret_cast<const v8bf*>(g + 24);
  *reinterpret_cast<v8bf*>(l)      = t0;
  *reinterpret_cast<v8bf*>(l + 8)  = t1;
  *reinterpret_cast<v8bf*>(l + 16) = t2;
  *reinterpret_cast<v8bf*>(l + 24) = t3;
#endif
}

__device__ __forceinline__ void stage_wait() {
#if HAVE_ASYNC_LDS
  __builtin_amdgcn_s_wait_asynccnt(0);
#endif
  // Fallback path: ds_store completion is covered by the barrier's dscnt wait.
}

// ---------------------------------------------------------------------------
// fp32 -> bf16 convert
// ---------------------------------------------------------------------------
__global__ void cvt_f32_bf16(const float* __restrict__ in, bf16* __restrict__ out, int n) {
  int i = blockIdx.x * 256 + threadIdx.x;
  if (i < n) out[i] = (bf16)in[i];
}

// ---------------------------------------------------------------------------
// QKV projection: Y = X @ W^T.  grid = (1024 m-tiles, 3 matrices), block 512.
// Wave w (0..15), subtile p (0..3) -> cols w*16 + p*256 (16x1024 per block).
// z==2 (V) is stored transposed: vt[b*1024 + e][s].
// ---------------------------------------------------------------------------
__global__ __launch_bounds__(512) void proj_qkv(
    const bf16* __restrict__ xb, const bf16* __restrict__ wb,
    bf16* __restrict__ qb, bf16* __restrict__ kb, bf16* __restrict__ vt) {
  __shared__ bf16 atile[16 * 1024];   // 32 KB
  const int tid  = threadIdx.x;
  const int wave = tid >> 5;
  const int lane = tid & 31;
  const int m0 = blockIdx.x * 16;
  const int which = blockIdx.y;                 // 0=q 1=k 2=v

  stage_panel(xb + (size_t)m0 * kD, kD, atile, tid);
  stage_wait();
  __syncthreads();

  const bf16* W = wb + (size_t)which * kD * kD;
  const bf16* brow[4];
#pragma unroll
  for (int p = 0; p < 4; ++p)
    brow[p] = W + (size_t)(wave * 16 + p * 256 + (lane & 15)) * kD;

  v8f acc[4] = {};
  for (int k0 = 0; k0 < kD; k0 += 32) {
    v16bf a = frag_lds(atile, k0, lane);
#pragma unroll
    for (int p = 0; p < 4; ++p) {
      v16bf bf = frag16x32(brow[p], k0, lane);
      acc[p] = wmma_bf16(a, bf, acc[p]);
    }
  }

  const int hi = lane >> 4;
  if (which == 2) {
    const int bidx = m0 >> 12;
#pragma unroll
    for (int p = 0; p < 4; ++p) {
      const int n = wave * 16 + p * 256 + (lane & 15);
#pragma unroll
      for (int r = 0; r < 8; ++r) {
        int m = m0 + r + 8 * hi;
        vt[((size_t)(bidx * kD + n)) * kS + (m & (kS - 1))] = (bf16)acc[p][r];
      }
    }
  } else {
    bf16* dst = (which == 0) ? qb : kb;
#pragma unroll
    for (int p = 0; p < 4; ++p) {
      const int n = wave * 16 + p * 256 + (lane & 15);
#pragma unroll
      for (int r = 0; r < 8; ++r) {
        int m = m0 + r + 8 * hi;
        dst[(size_t)m * kD + n] = (bf16)acc[p][r];
      }
    }
  }
}

// ---------------------------------------------------------------------------
// scores[b,q,kk] = sum_d q[b,q,d] * k[b,kk,d]   (bf16 out)
// grid = (256 q-tiles, 4 col-strips of 1024, 4 batches), block 512.
// q-panel (16x1024) async-staged once; k rows streamed direct.
// ---------------------------------------------------------------------------
__global__ __launch_bounds__(512) void scores_qkT(
    const bf16* __restrict__ qb, const bf16* __restrict__ kbuf, bf16* __restrict__ sc) {
  __shared__ bf16 atile[16 * 1024];   // 32 KB
  const int tid  = threadIdx.x;
  const int wave = tid >> 5;
  const int lane = tid & 31;
  const int b  = blockIdx.z;
  const int m0 = blockIdx.x * 16;                      // q row in batch
  const int nbase = blockIdx.y * 1024 + wave * 16;     // col strip base

  stage_panel(qb + (size_t)(b * kS + m0) * kD, kD, atile, tid);
  stage_wait();
  __syncthreads();

  const bf16* brow[4];
#pragma unroll
  for (int p = 0; p < 4; ++p)
    brow[p] = kbuf + (size_t)(b * kS + nbase + p * 256 + (lane & 15)) * kD;

  v8f acc[4] = {};
  for (int k0 = 0; k0 < kD; k0 += 32) {
    v16bf a = frag_lds(atile, k0, lane);
#pragma unroll
    for (int p = 0; p < 4; ++p) {
      v16bf bf = frag16x32(brow[p], k0, lane);
      acc[p] = wmma_bf16(a, bf, acc[p]);
    }
  }

  const int hi = lane >> 4;
#pragma unroll
  for (int p = 0; p < 4; ++p) {
    const int n = nbase + p * 256 + (lane & 15);
#pragma unroll
    for (int r = 0; r < 8; ++r) {
      int m = m0 + r + 8 * hi;
      sc[((size_t)b * kS + m) * kS + n] = (bf16)acc[p][r];
    }
  }
}

// ---------------------------------------------------------------------------
// Row softmax stats: max & sum(exp(s-max)) per row of 4096. One block per row.
// ---------------------------------------------------------------------------
__global__ void softmax_stats(const bf16* __restrict__ sc, float* __restrict__ stats) {
  __shared__ float red[256];
  const int tid = threadIdx.x;
  const size_t row = blockIdx.x;
  const bf16* p = sc + row * (size_t)kS;

  float mx = -3.0e38f;
  for (int i = tid; i < kS; i += 256) mx = fmaxf(mx, (float)p[i]);
  red[tid] = mx;
  __syncthreads();
  for (int s = 128; s > 0; s >>= 1) {
    if (tid < s) red[tid] = fmaxf(red[tid], red[tid + s]);
    __syncthreads();
  }
  mx = red[0];
  __syncthreads();

  float sum = 0.f;
  for (int i = tid; i < kS; i += 256) sum += __expf((float)p[i] - mx);
  red[tid] = sum;
  __syncthreads();
  for (int s = 128; s > 0; s >>= 1) {
    if (tid < s) red[tid] += red[tid + s];
    __syncthreads();
  }
  if (tid == 0) { stats[row * 2] = mx; stats[row * 2 + 1] = red[0]; }
}

// ---------------------------------------------------------------------------
// In-place softmax apply: P = exp(s - max) / sum  (bf16 elementwise).
// ---------------------------------------------------------------------------
__global__ void softmax_apply(bf16* __restrict__ sc, const float* __restrict__ stats) {
  const size_t i = (size_t)blockIdx.x * 256 + threadIdx.x;   // < 4*4096*4096
  const size_t row = i >> 12;
  const float mx  = stats[row * 2];
  const float inv = 1.0f / stats[row * 2 + 1];
  float v = __expf((float)sc[i] - mx) * inv;
  sc[i] = (bf16)v;
}

// ---------------------------------------------------------------------------
// attended = P @ V, fused with y = attended + x and y /= ||y||_2.
// One 512-thread block per 16 output rows covering ALL 1024 columns.
// K=4096 handled as 4 chunks: each 16x1024 P-chunk async-staged to LDS,
// inner 32-step loop barrier-free; v^T rows streamed direct (contiguous).
// Row sum-of-squares via lane shuffles + ds_add_f32, rsqrt scale on store.
// ---------------------------------------------------------------------------
__global__ __launch_bounds__(512) void attn_out(
    const bf16* __restrict__ sc, const bf16* __restrict__ vt,
    const float* __restrict__ x, float* __restrict__ out) {
  __shared__ bf16 atile[16 * 1024];   // 32 KB
  __shared__ float rowss[16];
  const int tid  = threadIdx.x;
  const int wave = tid >> 5;
  const int lane = tid & 31;
  const int m0 = blockIdx.x * 16;        // global row (0..16383)
  const int b  = m0 >> 12;               // batch

  if (tid < 16) rowss[tid] = 0.f;

  const bf16* abase = sc + (size_t)m0 * kS;     // P rows (bf16)
  const bf16* brow[4];
#pragma unroll
  for (int p = 0; p < 4; ++p)
    brow[p] = vt + (size_t)(b * kD + wave * 16 + p * 256 + (lane & 15)) * kS;

  v8f acc[4] = {};
  for (int c = 0; c < 4; ++c) {
    __syncthreads();                               // prior chunk fully consumed
    stage_panel(abase + c * 1024, kS, atile, tid);
    stage_wait();
    __syncthreads();                               // panel visible to all waves
    const int kbase = c * 1024;
    for (int kk = 0; kk < 1024; kk += 32) {
      v16bf a = frag_lds(atile, kk, lane);
#pragma unroll
      for (int p = 0; p < 4; ++p) {
        v16bf bf = frag16x32(brow[p], kbase + kk, lane);
        acc[p] = wmma_bf16(a, bf, acc[p]);
      }
    }
  }

  const int hi = lane >> 4;
  float part[8];
#pragma unroll
  for (int r = 0; r < 8; ++r) part[r] = 0.f;

#pragma unroll
  for (int p = 0; p < 4; ++p) {
    const int n = wave * 16 + p * 256 + (lane & 15);
#pragma unroll
    for (int r = 0; r < 8; ++r) {
      const int m = m0 + r + 8 * hi;
      float yv = acc[p][r] + x[(size_t)m * kD + n];
      acc[p][r] = yv;
      part[r] += yv * yv;
    }
  }
  // reduce across the 16 lanes sharing each row
#pragma unroll
  for (int r = 0; r < 8; ++r) {
#pragma unroll
    for (int msk = 1; msk < 16; msk <<= 1)
      part[r] += __shfl_xor(part[r], msk, 32);
  }
  if ((lane & 15) == 0) {
#pragma unroll
    for (int r = 0; r < 8; ++r) atomicAdd(&rowss[r + 8 * hi], part[r]);
  }
  __syncthreads();

  float inv[8];
#pragma unroll
  for (int r = 0; r < 8; ++r) inv[r] = rsqrtf(rowss[r + 8 * hi]);

#pragma unroll
  for (int p = 0; p < 4; ++p) {
    const int n = wave * 16 + p * 256 + (lane & 15);
#pragma unroll
    for (int r = 0; r < 8; ++r) {
      const int m = m0 + r + 8 * hi;
      out[(size_t)m * kD + n] = acc[p][r] * inv[r];
    }
  }
}

// ---------------------------------------------------------------------------
extern "C" void kernel_launch(void* const* d_in, const int* in_sizes, int n_in,
                              void* d_out, int out_size, void* d_ws, size_t ws_size,
                              hipStream_t stream) {
  (void)in_sizes; (void)n_in; (void)out_size; (void)ws_size;
  const float* x  = (const float*)d_in[0];
  const float* Wq = (const float*)d_in[1];
  const float* Wk = (const float*)d_in[2];
  const float* Wv = (const float*)d_in[3];
  float* out = (float*)d_out;

  char* ws = (char*)d_ws;
  bf16*  xb = (bf16*)(ws + XB_OFF);
  bf16*  wb = (bf16*)(ws + WB_OFF);
  bf16*  qb = (bf16*)(ws + QB_OFF);
  bf16*  kb = (bf16*)(ws + KB_OFF);
  bf16*  vt = (bf16*)(ws + VT_OFF);
  bf16*  sc = (bf16*)(ws + SC_OFF);
  float* st = (float*)(ws + ST_OFF);

  const int nX = kRows * kD;       // 16777216
  const int nW = kD * kD;          // 1048576

  cvt_f32_bf16<<<nX / 256, 256, 0, stream>>>(x,  xb, nX);
  cvt_f32_bf16<<<nW / 256, 256, 0, stream>>>(Wq, wb + (size_t)0 * nW, nW);
  cvt_f32_bf16<<<nW / 256, 256, 0, stream>>>(Wk, wb + (size_t)1 * nW, nW);
  cvt_f32_bf16<<<nW / 256, 256, 0, stream>>>(Wv, wb + (size_t)2 * nW, nW);

  proj_qkv<<<dim3(kRows / 16, 3), 512, 0, stream>>>(xb, wb, qb, kb, vt);

  scores_qkT<<<dim3(kS / 16, kS / 1024, kB), 512, 0, stream>>>(qb, kb, sc);

  softmax_stats<<<kRows, 256, 0, stream>>>(sc, st);

  softmax_apply<<<(unsigned)(((size_t)kB * kS * kS) / 256), 256, 0, stream>>>(sc, st);

  attn_out<<<kRows / 16, 512, 0, stream>>>(sc, vt, x, out);
}
